// _NonLocalNd_13099650253374
// MI455X (gfx1250) — matrix operand order repaired and checked by reading, not compiled, not run on hardware
//
#include <hip/hip_runtime.h>

typedef __attribute__((ext_vector_type(16))) _Float16 v16h;
typedef __attribute__((ext_vector_type(8)))  _Float16 v8h;
typedef __attribute__((ext_vector_type(8)))  float    v8f;
typedef unsigned int u32x4 __attribute__((ext_vector_type(4)));
typedef int          i32x8 __attribute__((ext_vector_type(8)));
typedef int          i32x4 __attribute__((ext_vector_type(4)));
typedef _Float16 h16;

#define L_TOT 9216
#define CIN   128
#define CI    64
#define NB    2
#define NCH   (L_TOT / 64)

#if __has_builtin(__builtin_amdgcn_tensor_load_to_lds)
#define USE_TDM 1
#endif

__device__ inline v16h cat16(v8h lo, v8h hi) {
  v16h r;
#pragma unroll
  for (int i = 0; i < 8; ++i) { r[i] = lo[i]; r[i + 8] = hi[i]; }
  return r;
}

__device__ inline v8f wmma16(v16h a, v16h b, v8f c) {
  return __builtin_amdgcn_wmma_f32_16x16x32_f16(false, a, false, b, (short)0, c, false, false);
}

#ifdef USE_TDM
// 2D tile DMA: global (row-major, stride0 elements of 2B) -> LDS, with
// 16B padding appended after every 128B row (pad_interval=4 -> 32 DWORDs,
// pad_amount=3 -> 4 DWORDs), matching LDS row stride of 72 f16.
__device__ inline void tdm_load_2d(unsigned lds_addr, const void* gptr,
                                   unsigned tensor_d0, unsigned tensor_d1,
                                   unsigned tile_d0, unsigned tile_d1,
                                   unsigned stride0) {
  unsigned long long ga = (unsigned long long)(size_t)gptr;
  u32x4 g0;
  g0[0] = 1u;                                    // count=1, user descriptor
  g0[1] = lds_addr;                              // LDS byte address
  g0[2] = (unsigned)ga;                          // global addr [31:0]
  g0[3] = (unsigned)((ga >> 32) & 0x1FFFFFFu) | (2u << 30);  // [56:32] | type=2
  i32x8 g1;
  g1[0] = (1 << 16) | (1 << 20) | (4 << 22) | (3 << 25);  // data_size=2B, pad on
  g1[1] = (int)((tensor_d0 & 0xFFFFu) << 16);             // tensor_dim0 lo16
  g1[2] = (int)((tensor_d0 >> 16) | ((tensor_d1 & 0xFFFFu) << 16));
  g1[3] = (int)((tensor_d1 >> 16) | (tile_d0 << 16));
  g1[4] = (int)tile_d1;                                   // tile_dim2 = 0
  g1[5] = (int)stride0;                                   // dim0 stride lo32
  g1[6] = 0;                                              // stride0 hi / stride1 lo
  g1[7] = 0;
  i32x4 g2 = {1, 1, 0, 0};          // tensor_dim2=1, tensor_dim3=1
  i32x4 g3 = {0, 1 << 16, 0, 0};    // tensor_dim4=1
  i32x8 g4 = {0, 0, 0, 0, 0, 0, 0, 0};   // 6-arg toolchain: extra group, zeroed
  __builtin_amdgcn_tensor_load_to_lds(g0, g1, g2, g3, g4, 0);
}
#endif

// ---------------------------------------------------------------------------
// Kernel 1: fused theta/phi/g projections: 1x1 conv + BN + ReLU
//   thT [n][L][64] f16, phT [n][L][64] f16, gX [n][64][L] f16
// block = 128 threads (4 waves); each block owns 128 spatial positions.
// wave w owns output-channel rows 16w..16w+15 of all three projections.
// ---------------------------------------------------------------------------
__global__ __launch_bounds__(128) void nl_proj_kernel(
    const float* __restrict__ x,
    const float* __restrict__ gW, const float* __restrict__ gG,
    const float* __restrict__ gB, const float* __restrict__ gM,
    const float* __restrict__ gV,
    const float* __restrict__ tW, const float* __restrict__ tG,
    const float* __restrict__ tB, const float* __restrict__ tM,
    const float* __restrict__ tV,
    const float* __restrict__ pW, const float* __restrict__ pG,
    const float* __restrict__ pB, const float* __restrict__ pM,
    const float* __restrict__ pV,
    h16* __restrict__ thT, h16* __restrict__ phT, h16* __restrict__ gX)
{
  __shared__ h16   xt[128 * 136];   // [l][cin], padded to 136
  __shared__ float sc[192], bi[192];

  const int tid  = threadIdx.x;
  const int wave = tid >> 5;
  const int lane = tid & 31;
  const int hlf  = lane >> 4;
  const int ln16 = lane & 15;

  const int n  = blockIdx.x / (L_TOT / 128);
  const int l0 = (blockIdx.x % (L_TOT / 128)) * 128;

  // BN folded constants, stacked channels: 0-63 theta, 64-127 phi, 128-191 g
  for (int ch = tid; ch < 192; ch += 128) {
    int p = ch >> 6, c = ch & 63;
    const float* Gv = (p == 0) ? tG : (p == 1) ? pG : gG;
    const float* Bv = (p == 0) ? tB : (p == 1) ? pB : gB;
    const float* Mv = (p == 0) ? tM : (p == 1) ? pM : gM;
    const float* Vv = (p == 0) ? tV : (p == 1) ? pV : gV;
    float s = Gv[c] * rsqrtf(Vv[c] + 1e-5f);
    sc[ch] = s;
    bi[ch] = Bv[c] - Mv[c] * s;
  }

  // load x tile transposed into LDS as f16 [l][cin]
  {
    const float* xb = x + (size_t)n * CIN * L_TOT + l0 + tid;
#pragma unroll 8
    for (int c = 0; c < CIN; ++c)
      xt[tid * 136 + c] = (h16)xb[(size_t)c * L_TOT];
  }
  __syncthreads();

  // A-fragments of the weights; t=0 theta, t=1 phi, t=2 g (static!)
  v16h wa[3][4];
#pragma unroll
  for (int t = 0; t < 3; ++t) {
    const float* Wp = (t == 0) ? tW : (t == 1) ? pW : gW;
    const float* wr = Wp + (size_t)(wave * 16 + ln16) * CIN;
#pragma unroll
    for (int ks = 0; ks < 4; ++ks) {
      int d0 = ks * 32 + 8 * hlf;
      v16h a;
#pragma unroll
      for (int i = 0; i < 8; ++i) {
        a[i]     = (h16)wr[d0 + i];
        a[i + 8] = (h16)wr[d0 + 16 + i];
      }
      wa[t][ks] = a;
    }
  }

  for (int nt = 0; nt < 8; ++nt) {
    const h16* xr = &xt[(nt * 16 + ln16) * 136];
    v8f c0 = {0,0,0,0,0,0,0,0}, c1 = c0, c2 = c0;
#pragma unroll
    for (int ks = 0; ks < 4; ++ks) {
      v8h b0 = *(const v8h*)(xr + ks * 32 + 8 * hlf);
      v8h b1 = *(const v8h*)(xr + ks * 32 + 8 * hlf + 16);
      v16h b = cat16(b0, b1);
      c0 = wmma16(wa[0][ks], b, c0);
      c1 = wmma16(wa[1][ks], b, c1);
      c2 = wmma16(wa[2][ks], b, c2);
    }
    const int lcol = l0 + nt * 16 + ln16;

    // theta / phi -> [l][64] layout, 8 consecutive d per lane -> b128 store
#pragma unroll
    for (int t = 0; t < 2; ++t) {
      v8f cc = t ? c1 : c0;
      h16* dst = (t ? phT : thT) +
                 ((size_t)n * L_TOT + lcol) * 64 + wave * 16 + 8 * hlf;
      v8h hv;
#pragma unroll
      for (int r = 0; r < 8; ++r) {
        int ch = wave * 16 + t * 64 + r + 8 * hlf;
        float v = cc[r] * sc[ch] + bi[ch];
        hv[r] = (h16)(v > 0.f ? v : 0.f);
      }
      *(v8h*)dst = hv;
    }
    // g -> [d][L] layout
    {
      h16* dst = gX + ((size_t)n * CI + wave * 16 + 8 * hlf) * L_TOT + lcol;
#pragma unroll
      for (int r = 0; r < 8; ++r) {
        int ch = wave * 16 + 128 + r + 8 * hlf;
        float v = c2[r] * sc[ch] + bi[ch];
        dst[(size_t)r * L_TOT] = (h16)(v > 0.f ? v : 0.f);
      }
    }
  }
}

// ---------------------------------------------------------------------------
// Kernel 2: flash attention (online softmax) + out-proj + BN + ReLU + residual
// block = 128 threads (4 waves); each wave owns a 16-row query tile (64/WG).
// Keys stream in 64-wide chunks via TDM (double-buffered LDS tiles).
// ---------------------------------------------------------------------------
__global__ __launch_bounds__(128) void nl_attn_kernel(
    const h16* __restrict__ thT, const h16* __restrict__ phT,
    const h16* __restrict__ gX,  const float* __restrict__ x,
    const float* __restrict__ oW, const float* __restrict__ oG,
    const float* __restrict__ oB, const float* __restrict__ oM,
    const float* __restrict__ oV, float* __restrict__ out)
{
  __shared__ h16 kT[2][64 * 72];     // [key][d], double buffered
  __shared__ h16 gT[2][64 * 72];     // [d][key], double buffered
  __shared__ h16 pT[4][16 * 72];     // per-wave P tile [row][key]
  __shared__ h16 yT[64 * 72];        // [l_local][d]
  __shared__ float osc[128], obi[128];

  const int tid  = threadIdx.x;
  const int wave = tid >> 5;
  const int lane = tid & 31;
  const int hlf  = lane >> 4;
  const int ln16 = lane & 15;

  const int n  = blockIdx.x / NCH;
  const int q0 = (blockIdx.x % NCH) * 64;

  const h16* phN = phT + (size_t)n * L_TOT * CI;   // [L][64]
  const h16* gN  = gX  + (size_t)n * CI * L_TOT;   // [64][L]

  {
    float s = oG[tid] * rsqrtf(oV[tid] + 1e-5f);
    osc[tid] = s;
    obi[tid] = oB[tid] - oM[tid] * s;
  }

  // Q fragments (A layout), 2 k-steps over d=64
  v16h qa[2];
  {
    const h16* qrow = thT + ((size_t)n * L_TOT + q0 + wave * 16 + ln16) * 64;
#pragma unroll
    for (int ks = 0; ks < 2; ++ks) {
      v8h a0 = *(const v8h*)(qrow + ks * 32 + 8 * hlf);
      v8h a1 = *(const v8h*)(qrow + ks * 32 + 8 * hlf + 16);
      qa[ks] = cat16(a0, a1);
    }
  }

  v8f acc[4];
  float mrow[8], lrow[8];
#pragma unroll
  for (int nt = 0; nt < 4; ++nt)
#pragma unroll
    for (int i = 0; i < 8; ++i) acc[nt][i] = 0.f;
#pragma unroll
  for (int r = 0; r < 8; ++r) { mrow[r] = -1e30f; lrow[r] = 0.f; }

#ifdef USE_TDM
  const unsigned kT_lds = (unsigned)(size_t)(void*)&kT[0][0];
  const unsigned gT_lds = (unsigned)(size_t)(void*)&gT[0][0];
  if (wave == 0) {
    tdm_load_2d(kT_lds, phN, 64, L_TOT, 64, 64, 64);
    tdm_load_2d(gT_lds, gN,  L_TOT, 64, 64, 64, L_TOT);
  }
#else
  {
    int r = tid >> 1, seg = tid & 1;
    const v8h* s1 = (const v8h*)(phN + (size_t)r * 64 + seg * 32);
    v8h* d1 = (v8h*)(&kT[0][0] + r * 72 + seg * 32);
    const v8h* s2 = (const v8h*)(gN + (size_t)r * L_TOT + seg * 32);
    v8h* d2 = (v8h*)(&gT[0][0] + r * 72 + seg * 32);
#pragma unroll
    for (int j = 0; j < 4; ++j) { d1[j] = s1[j]; d2[j] = s2[j]; }
  }
#endif

  for (int i = 0; i < NCH; ++i) {
    const int cur = i & 1;
#ifdef USE_TDM
    if (wave == 0) __builtin_amdgcn_s_wait_tensorcnt(0);
#endif
    __syncthreads();   // tiles[cur] ready; tiles[cur^1] fully consumed

    if (i + 1 < NCH) {
      const int kc = (i + 1) * 64;
#ifdef USE_TDM
      if (wave == 0) {
        tdm_load_2d(kT_lds + (cur ^ 1) * 9216, phN + (size_t)kc * 64,
                    64, L_TOT, 64, 64, 64);
        tdm_load_2d(gT_lds + (cur ^ 1) * 9216, gN + kc,
                    L_TOT, 64, 64, 64, L_TOT);
      }
#else
      int r = tid >> 1, seg = tid & 1;
      const v8h* s1 = (const v8h*)(phN + ((size_t)kc + r) * 64 + seg * 32);
      v8h* d1 = (v8h*)(&kT[cur ^ 1][0] + r * 72 + seg * 32);
      const v8h* s2 = (const v8h*)(gN + (size_t)r * L_TOT + kc + seg * 32);
      v8h* d2 = (v8h*)(&gT[cur ^ 1][0] + r * 72 + seg * 32);
#pragma unroll
      for (int j = 0; j < 4; ++j) { d1[j] = s1[j]; d2[j] = s2[j]; }
#endif
    }

    // S = (Q K^T) * 1/sqrt(64)
    v8f s[4];
#pragma unroll
    for (int nt = 0; nt < 4; ++nt) {
      v8f c = {0,0,0,0,0,0,0,0};
#pragma unroll
      for (int ks = 0; ks < 2; ++ks) {
        const h16* kr = &kT[cur][(nt * 16 + ln16) * 72 + ks * 32 + 8 * hlf];
        v16h b = cat16(*(const v8h*)kr, *(const v8h*)(kr + 16));
        c = wmma16(qa[ks], b, c);
      }
#pragma unroll
      for (int j = 0; j < 8; ++j) c[j] *= 0.125f;
      s[nt] = c;
    }

    // online softmax
    float mnew[8], corr[8];
#pragma unroll
    for (int r = 0; r < 8; ++r) {
      float mx = fmaxf(fmaxf(s[0][r], s[1][r]), fmaxf(s[2][r], s[3][r]));
      mx = fmaxf(mx, __shfl_xor(mx, 1, 32));
      mx = fmaxf(mx, __shfl_xor(mx, 2, 32));
      mx = fmaxf(mx, __shfl_xor(mx, 4, 32));
      mx = fmaxf(mx, __shfl_xor(mx, 8, 32));
      mnew[r] = fmaxf(mrow[r], mx);
      corr[r] = __expf(mrow[r] - mnew[r]);
    }
#pragma unroll
    for (int nt = 0; nt < 4; ++nt)
#pragma unroll
      for (int r = 0; r < 8; ++r) {
        acc[nt][r] *= corr[r];
        s[nt][r] = __expf(s[nt][r] - mnew[r]);     // P values
      }
#pragma unroll
    for (int r = 0; r < 8; ++r) {
      float su = (s[0][r] + s[1][r]) + (s[2][r] + s[3][r]);
      su += __shfl_xor(su, 1, 32);
      su += __shfl_xor(su, 2, 32);
      su += __shfl_xor(su, 4, 32);
      su += __shfl_xor(su, 8, 32);
      lrow[r] = lrow[r] * corr[r] + su;
      mrow[r] = mnew[r];
    }

    // store P (C layout -> [row][key] in LDS)
#pragma unroll
    for (int nt = 0; nt < 4; ++nt)
#pragma unroll
      for (int r = 0; r < 8; ++r)
        pT[wave][(r + 8 * hlf) * 72 + nt * 16 + ln16] = (h16)s[nt][r];
    __syncthreads();   // P visible / DS stores complete

    // acc += P @ G
    const h16* prow = &pT[wave][ln16 * 72];
#pragma unroll
    for (int ks = 0; ks < 2; ++ks) {
      v8h a0 = *(const v8h*)(prow + ks * 32 + 8 * hlf);
      v8h a1 = *(const v8h*)(prow + ks * 32 + 8 * hlf + 16);
      v16h pa = cat16(a0, a1);
#pragma unroll
      for (int nt = 0; nt < 4; ++nt) {
        const h16* gr = &gT[cur][(nt * 16 + ln16) * 72 + ks * 32 + 8 * hlf];
        v16h b = cat16(*(const v8h*)gr, *(const v8h*)(gr + 16));
        acc[nt] = wmma16(pa, b, acc[nt]);
      }
    }
  }

  // normalize and stage y into LDS as [l_local][d]
  __syncthreads();
#pragma unroll
  for (int nt = 0; nt < 4; ++nt)
#pragma unroll
    for (int r = 0; r < 8; ++r)
      yT[(wave * 16 + r + 8 * hlf) * 72 + nt * 16 + ln16] =
          (h16)(acc[nt][r] / lrow[r]);
  __syncthreads();

  // out projection (128x64 @ 64x64) + BN + ReLU + residual
#pragma unroll
  for (int t = 0; t < 2; ++t) {
    const int cbase = (wave * 2 + t) * 16;
    const float* wr = oW + (size_t)(cbase + ln16) * CI;
    v16h wa2[2];
#pragma unroll
    for (int ks = 0; ks < 2; ++ks) {
      int d0 = ks * 32 + 8 * hlf;
      v16h a;
#pragma unroll
      for (int i = 0; i < 8; ++i) {
        a[i]     = (h16)wr[d0 + i];
        a[i + 8] = (h16)wr[d0 + 16 + i];
      }
      wa2[ks] = a;
    }
#pragma unroll
    for (int nt = 0; nt < 4; ++nt) {
      v8f c = {0,0,0,0,0,0,0,0};
#pragma unroll
      for (int ks = 0; ks < 2; ++ks) {
        const h16* yr = &yT[(nt * 16 + ln16) * 72 + ks * 32 + 8 * hlf];
        v16h b = cat16(*(const v8h*)yr, *(const v8h*)(yr + 16));
        c = wmma16(wa2[ks], b, c);
      }
      const int lcol = q0 + nt * 16 + ln16;
      const float* xr   = x   + ((size_t)n * CIN + cbase + 8 * hlf) * L_TOT + lcol;
      float*       orow = out + ((size_t)n * CIN + cbase + 8 * hlf) * L_TOT + lcol;
#pragma unroll
      for (int r = 0; r < 8; ++r) {
        int ch = cbase + r + 8 * hlf;
        float v = c[r] * osc[ch] + obi[ch];
        v = v > 0.f ? v : 0.f;
        orow[(size_t)r * L_TOT] = xr[(size_t)r * L_TOT] + v;
      }
    }
  }
}

extern "C" void kernel_launch(void* const* d_in, const int* in_sizes, int n_in,
                              void* d_out, int out_size, void* d_ws, size_t ws_size,
                              hipStream_t stream) {
  const float* x  = (const float*)d_in[0];
  const float* gW = (const float*)d_in[1];
  const float* gG = (const float*)d_in[2];
  const float* gB = (const float*)d_in[3];
  const float* gM = (const float*)d_in[4];
  const float* gV = (const float*)d_in[5];
  const float* tW = (const float*)d_in[6];
  const float* tG = (const float*)d_in[7];
  const float* tB = (const float*)d_in[8];
  const float* tM = (const float*)d_in[9];
  const float* tV = (const float*)d_in[10];
  const float* pW = (const float*)d_in[11];
  const float* pG = (const float*)d_in[12];
  const float* pB = (const float*)d_in[13];
  const float* pM = (const float*)d_in[14];
  const float* pV = (const float*)d_in[15];
  const float* oW = (const float*)d_in[16];
  const float* oG = (const float*)d_in[17];
  const float* oB = (const float*)d_in[18];
  const float* oM = (const float*)d_in[19];
  const float* oV = (const float*)d_in[20];

  const size_t per = (size_t)NB * L_TOT * CI;   // elements per f16 scratch tensor
  h16* thT = (h16*)d_ws;
  h16* phT = thT + per;
  h16* gX  = phT + per;

  nl_proj_kernel<<<dim3(NB * (L_TOT / 128)), dim3(128), 0, stream>>>(
      x, gW, gG, gB, gM, gV, tW, tG, tB, tM, tV, pW, pG, pB, pM, pV,
      thT, phT, gX);

  nl_attn_kernel<<<dim3(NB * NCH), dim3(128), 0, stream>>>(
      thT, phT, gX, x, oW, oG, oB, oM, oV, (float*)d_out);
}
